// PrecursorClassifier_69114613730634
// MI455X (gfx1250) — matrix operand order repaired and checked by reading, not compiled, not run on hardware
//
#include <hip/hip_runtime.h>
#include <hip/hip_fp16.h>
#include <hip/hip_bf16.h>

typedef __attribute__((ext_vector_type(16))) _Float16 v16h;
typedef __attribute__((ext_vector_type(8)))  float    v8f;

union Frag { unsigned u[8]; v16h h; };
union H2U  { _Float16 h[2]; unsigned u; };

#define C_CRY 8000
#define NN    40000      // nodes
#define MM    200000     // edges

struct HeadP {
  const unsigned *w1g;   // gate hidden W packed (din x 256)
  const unsigned *wg;    // gate out vec packed (256 -> 128 dwords)
  const unsigned *w1m;   // msg hidden W packed (din x 256)
  const unsigned *wm;    // msg out W packed (256 x 64)
  const float *b1g, *bg, *b1m, *bm, *pw;
};
struct HeadP3 { HeadP h[3]; };

// ---------------- weight packing into WMMA B-fragment order ----------------
// out[ ((kc*ntiles + nb)*32 + lane)*8 + j ] = pack2( W[K][col], W[K+1][col] )
// col = nb*16 + (lane&15); K = 32*kc + (lane<16?0:16) + 2*j   (contiguous k order, matches A loads)
__global__ void pack_kernel(const float* __restrict__ W, int dout, int total,
                            unsigned* __restrict__ out) {
  int t = blockIdx.x * blockDim.x + threadIdx.x;
  if (t >= total) return;
  int j    = t & 7;
  int lane = (t >> 3) & 31;
  int rest = t >> 8;
  int ntiles = dout >> 4;
  int nb = rest % ntiles;
  int kc = rest / ntiles;
  int col = nb * 16 + (lane & 15);
  int K = kc * 32 + ((lane < 16) ? 0 : 16) + 2 * j;
  H2U p;
  p.h[0] = (_Float16)W[(size_t)K * dout + col];
  p.h[1] = (_Float16)W[(size_t)(K + 1) * dout + col];
  out[t] = p.u;
}

// gate output vector (256x1) -> 128 packed K-pairs
__global__ void packg_kernel(const float* __restrict__ W, unsigned* __restrict__ out) {
  int t = threadIdx.x;  // 128 threads
  H2U p;
  p.h[0] = (_Float16)W[2 * t];
  p.h[1] = (_Float16)W[2 * t + 1];
  out[t] = p.u;
}

// ---------------- embedding: x[:, :63] = ef @ W + b ; x[:,63] = ew ----------
__global__ void embed_kernel(const float* __restrict__ ef, const float* __restrict__ eW,
                             const float* __restrict__ eb, const float* __restrict__ ew,
                             float* __restrict__ xA) {
  __shared__ float row[200];
  const int n = blockIdx.x, t = threadIdx.x;  // 64 threads
  for (int i = t; i < 200; i += 64) row[i] = ef[(size_t)n * 200 + i];
  __syncthreads();
  if (t < 63) {
    float a = eb[t];
    for (int k = 0; k < 200; ++k) a += row[k] * eW[(size_t)k * 63 + t];
    xA[(size_t)n * 64 + t] = a;
  } else if (t == 63) {
    xA[(size_t)n * 64 + 63] = ew[n];
  }
}

__global__ void f2h_kernel(const float* __restrict__ x, _Float16* __restrict__ xh, int n) {
  int i = blockIdx.x * blockDim.x + threadIdx.x;
  if (i < n) xh[i] = (_Float16)x[i];
}

// source_elem_fea: mean of first two nodes per crystal
__global__ void source_kernel(const float* __restrict__ x, float* __restrict__ out) {
  int i = blockIdx.x * blockDim.x + threadIdx.x;
  if (i >= C_CRY * 64) return;
  int c = i >> 6, col = i & 63;
  out[i] = 0.5f * (x[(size_t)(c * 5) * 64 + col] + x[(size_t)(c * 5 + 1) * 64 + col]);
}

// ---------------- WMMA helpers ----------------
__device__ __forceinline__ v8f wmma16(const Frag& a, const Frag& b, v8f c) {
  return __builtin_amdgcn_wmma_f32_16x16x32_f16(false, a.h, false, b.h, (short)0, c,
                                                false, false);
}

// hidden = LeakyReLU(fea @ W1 + b1), staged to LDS as row-major 16x256 f16
template <int DINC>
__device__ __forceinline__ void hidden_pass(const Frag* A, const unsigned* __restrict__ w1,
                                            const float* __restrict__ b1, _Float16* sh,
                                            int lane) {
  const int c = lane & 15;
  const int rb = (lane < 16) ? 0 : 8;
#pragma unroll
  for (int nb = 0; nb < 16; ++nb) {
    float bc = b1[nb * 16 + c];
    v8f acc;
#pragma unroll
    for (int j = 0; j < 8; ++j) acc[j] = bc;
#pragma unroll
    for (int kc = 0; kc < DINC; ++kc) {
      Frag B;
      const unsigned* bp = w1 + ((((kc * 16 + nb) * 32) + lane) << 3);
#pragma unroll
      for (int j = 0; j < 8; ++j) B.u[j] = bp[j];
      acc = wmma16(A[kc], B, acc);
    }
#pragma unroll
    for (int j = 0; j < 8; ++j) {
      float v = acc[j];
      v = (v > 0.f) ? v : 0.01f * v;  // LeakyReLU(0.01)
      sh[(j + rb) * 256 + nb * 16 + c] = (_Float16)v;
    }
  }
}

// reload staged hidden (16x256) as 8 WMMA A-fragments (K chunks of 32)
__device__ __forceinline__ void load_hfrags(Frag* HA, const _Float16* sh, int lane) {
  const int offs[8] = {0, 1, 2, 3, 4, 5, 6, 7};
  const int add = (lane < 16) ? 0 : 8;
  const unsigned* hu = (const unsigned*)(sh + (size_t)(lane & 15) * 256);
#pragma unroll
  for (int kc = 0; kc < 8; ++kc)
#pragma unroll
    for (int j = 0; j < 8; ++j) HA[kc].u[j] = hu[16 * kc + offs[j] + add];
}

// ---------------- fused attention head (edge message / crystal pooling) ----
// 160 threads = 5 waves; each wave: 16 rows; block: 80 rows = 16 segments of 5.
template <int DINC, bool EDGE>
__global__ __launch_bounds__(160) void head_kernel(
    const _Float16* __restrict__ xh, const float* __restrict__ ew,
    const int* __restrict__ selfIdx, const int* __restrict__ nbrIdx, HeadP3 hp3,
    float* __restrict__ outp) {
  __shared__ _Float16 sHid[5][16 * 256];  // 40 KB: per-wave hidden staging
  __shared__ float sMsg[80][64];          // 20 KB: msg per row
  __shared__ float sGate[80];
  __shared__ float sW[80];

  const int w = threadIdx.x >> 5, lane = threadIdx.x & 31;
  const HeadP hp = hp3.h[blockIdx.y];
  const int r0 = (blockIdx.x * 5 + w) * 16;
  const int rr = r0 + (lane & 15);
  const int offs[8] = {0, 1, 2, 3, 4, 5, 6, 7};
  const int add = (lane < 16) ? 0 : 8;

  // ---- build A fragments (row = edge or node, K = input features) ----
  Frag A[DINC];
  if constexpr (EDGE) {
    const unsigned* ps = (const unsigned*)(xh + (size_t)selfIdx[rr] * 64);
    const unsigned* pn = (const unsigned*)(xh + (size_t)nbrIdx[rr] * 64);
#pragma unroll
    for (int j = 0; j < 8; ++j) {
      int o = offs[j] + add;
      A[0].u[j] = ps[o];      A[1].u[j] = ps[16 + o];
      A[2].u[j] = pn[o];      A[3].u[j] = pn[16 + o];
    }
  } else {
    const unsigned* ps = (const unsigned*)(xh + (size_t)rr * 64);
#pragma unroll
    for (int j = 0; j < 8; ++j) {
      int o = offs[j] + add;
      A[0].u[j] = ps[o];      A[1].u[j] = ps[16 + o];
    }
  }

  // ---- gate network: hidden GEMM + gate dot (replicated-column WMMA) ----
  hidden_pass<DINC>(A, hp.w1g, hp.b1g, &sHid[w][0], lane);
  Frag HA[8];
  load_hfrags(HA, &sHid[w][0], lane);
  v8f accS;
#pragma unroll
  for (int j = 0; j < 8; ++j) accS[j] = 0.f;
#pragma unroll
  for (int kc = 0; kc < 8; ++kc) {
    Frag B;
    const unsigned* gp = hp.wg + 16 * kc + ((lane < 16) ? 0 : 8);
#pragma unroll
    for (int j = 0; j < 8; ++j) B.u[j] = gp[j];  // every column = gate vec
    accS = wmma16(HA[kc], B, accS);
  }
  if ((lane & 15) == 0) {
    float bg0 = hp.bg[0];
    int rb = (lane < 16) ? 0 : 8;
#pragma unroll
    for (int j = 0; j < 8; ++j) sGate[w * 16 + rb + j] = accS[j] + bg0;
  }

  // ---- message network: hidden GEMM + 256x64 output GEMM ----
  hidden_pass<DINC>(A, hp.w1m, hp.b1m, &sHid[w][0], lane);
  load_hfrags(HA, &sHid[w][0], lane);
#pragma unroll
  for (int nb = 0; nb < 4; ++nb) {
    float bc = hp.bm[nb * 16 + (lane & 15)];
    v8f acc;
#pragma unroll
    for (int j = 0; j < 8; ++j) acc[j] = bc;
#pragma unroll
    for (int kc = 0; kc < 8; ++kc) {
      Frag B;
      const unsigned* bp = hp.wm + ((((kc * 4 + nb) * 32) + lane) << 3);
#pragma unroll
      for (int j = 0; j < 8; ++j) B.u[j] = bp[j];
      acc = wmma16(HA[kc], B, acc);
    }
    int rb = (lane < 16) ? 0 : 8;
#pragma unroll
    for (int j = 0; j < 8; ++j) sMsg[w * 16 + rb + j][nb * 16 + (lane & 15)] = acc[j];
  }

  if (lane < 16) {
    float wv;
    if constexpr (EDGE) wv = ew[nbrIdx[r0 + lane]];
    else                wv = ew[r0 + lane];
    sW[w * 16 + lane] = wv;
  }
  __syncthreads();

  // ---- fused segment softmax pooling (5 rows / segment, 16 segments) ----
  const float pw = hp.pw[0];
  for (int o = threadIdx.x; o < 16 * 64; o += 160) {
    int nl = o >> 6, col = o & 63;
    float mx = -1e30f;
#pragma unroll
    for (int k = 0; k < 5; ++k) mx = fmaxf(mx, sGate[nl * 5 + k]);
    float s = 0.f, accv = 0.f;
#pragma unroll
    for (int k = 0; k < 5; ++k) {
      float e = powf(sW[nl * 5 + k], pw) * expf(sGate[nl * 5 + k] - mx);
      s += e;
      accv += e * sMsg[nl * 5 + k][col];
    }
    int seg = blockIdx.x * 16 + nl;
    atomicAdd(outp + (size_t)seg * 64 + col, (accv / (s + 1e-10f)) * (1.f / 3.f));
  }
}

// ---------------- host orchestration ----------------
// Assumed pytree leaf order (sorted dict keys): cry(3 heads) | embed.W, embed.b |
// graphs(3 layers x 3 heads) | elem_weights, elem_fea, self_idx, nbr_idx, ...
// Per head (keys gate<msg<pow; snet Wout<bout<hidden):
//   +0 gate.Wout +1 gate.bout +2 gate.hW +3 gate.hb
//   +4 msg.Wout  +5 msg.bout  +6 msg.hW  +7 msg.hb  +8 pow
extern "C" void kernel_launch(void* const* d_in, const int* in_sizes, int n_in,
                              void* d_out, int out_size, void* d_ws, size_t ws_size,
                              hipStream_t stream) {
  (void)in_sizes; (void)n_in; (void)out_size; (void)ws_size;
  char* ws = (char*)d_ws;
  float*    xA = (float*)ws;                          // 10.24 MB
  float*    xB = (float*)(ws + 10240000);             // 10.24 MB
  _Float16* xh = (_Float16*)(ws + 20480000);          // 5.12 MB
  unsigned* pk = (unsigned*)(ws + 25600000);          // packed weights

  auto F = [&](int i) { return (const float*)d_in[i]; };
  auto I = [&](int i) { return (const int*)d_in[i]; };
  const float* ew      = F(110);
  const float* ef      = F(111);
  const int*   selfIdx = I(112);
  const int*   nbrIdx  = I(113);

  auto packM = [&](const float* W, int din, int dout, unsigned* out) {
    int total = din * dout / 2;
    pack_kernel<<<(total + 255) / 256, 256, 0, stream>>>(W, dout, total, out);
  };

  // pack edge-head weights: per head stride 41088 dwords
  for (int e = 0; e < 9; ++e) {
    unsigned* base = pk + (size_t)e * 41088;
    int L = 29 + e * 9;
    packM(F(L + 2), 128, 256, base);            // gate hidden W
    packM(F(L + 6), 128, 256, base + 16384);    // msg hidden W
    packM(F(L + 4), 256, 64,  base + 32768);    // msg Wout
    packg_kernel<<<1, 128, 0, stream>>>(F(L + 0), base + 40960);
  }
  // pack crystal-head weights: stride 24704 dwords
  unsigned* cryPk = pk + (size_t)9 * 41088;
  for (int h = 0; h < 3; ++h) {
    unsigned* base = cryPk + (size_t)h * 24704;
    int L = h * 9;
    packM(F(L + 2), 64, 256, base);
    packM(F(L + 6), 64, 256, base + 8192);
    packM(F(L + 4), 256, 64, base + 16384);
    packg_kernel<<<1, 128, 0, stream>>>(F(L + 0), base + 24576);
  }

  // embedding
  embed_kernel<<<NN, 64, 0, stream>>>(ef, F(27), F(28), ew, xA);

  // 3 message-passing layers
  float* xcur = xA;
  float* xnext = xB;
  for (int g = 0; g < 3; ++g) {
    f2h_kernel<<<(NN * 64 + 255) / 256, 256, 0, stream>>>(xcur, xh, NN * 64);
    hipMemcpyAsync(xnext, xcur, (size_t)NN * 64 * 4, hipMemcpyDeviceToDevice, stream);
    HeadP3 hp{};
    for (int h = 0; h < 3; ++h) {
      int e = g * 3 + h;
      int L = 29 + e * 9;
      unsigned* base = pk + (size_t)e * 41088;
      hp.h[h] = {base, base + 40960, base + 16384, base + 32768,
                 F(L + 3), F(L + 1), F(L + 7), F(L + 5), F(L + 8)};
    }
    head_kernel<4, true><<<dim3(MM / 80, 3), 160, 0, stream>>>(xh, ew, selfIdx, nbrIdx,
                                                               hp, xnext);
    float* t = xcur; xcur = xnext; xnext = t;
  }

  // crystal pooling heads -> d_out[0 : C*64)
  hipMemsetAsync(d_out, 0, (size_t)C_CRY * 64 * 4, stream);
  f2h_kernel<<<(NN * 64 + 255) / 256, 256, 0, stream>>>(xcur, xh, NN * 64);
  HeadP3 hc{};
  for (int h = 0; h < 3; ++h) {
    int L = h * 9;
    unsigned* base = cryPk + (size_t)h * 24704;
    hc.h[h] = {base, base + 24576, base + 8192, base + 16384,
               F(L + 3), F(L + 1), F(L + 7), F(L + 5), F(L + 8)};
  }
  head_kernel<2, false><<<dim3(NN / 80, 3), 160, 0, stream>>>(xh, ew, nullptr, nullptr,
                                                              hc, (float*)d_out);
  // source element features -> d_out[C*64 : 2*C*64)
  source_kernel<<<(C_CRY * 64 + 255) / 256, 256, 0, stream>>>(xcur,
                                                              (float*)d_out + C_CRY * 64);
}
